// SparseConvNeXtBlock_13778255085701
// MI455X (gfx1250) — matrix operand order repaired and checked by reading, not compiled, not run on hardware
//
#include <hip/hip_runtime.h>
#include <math.h>

typedef __bf16 bf16;
typedef __attribute__((ext_vector_type(16))) __bf16 v16bf;
typedef __attribute__((ext_vector_type(8)))  float  v8f;

// ---------------- problem constants ----------------
constexpr int kDim  = 384;
constexpr int kGC   = 48;
constexpr int kHid  = 1536;
constexpr int kHW   = 1024;     // 32*32
constexpr int kNPix = 32768;    // 32 * 1024 pixels (M of both GEMMs)
constexpr int kKT1  = 12;       // DIM/32  K-tiles for GEMM1
constexpr int kKT2  = 48;       // HID/32  K-tiles for GEMM2

// ---------------- workspace layout (bytes) ----------------
constexpr size_t OFF_P1 = 0;                        // 12288 f32 per-block sum(y*m)
constexpr size_t OFF_P2 = OFF_P1 + 49152;           // 12288 f32 per-block sum(y*y*m)
constexpr size_t OFF_SA = OFF_P2 + 49152;           // 384 f32 BN scale a_c
constexpr size_t OFF_SB = OFF_SA + 1536;            // 384 f32 BN shift b_c
constexpr size_t OFF_W1 = OFF_SB + 1536;            // fc1 weights bf16, WMMA-B swizzled (1536*384)
constexpr size_t OFF_W2 = OFF_W1 + 1179648;         // fc2 weights bf16, WMMA-B swizzled (384*1536)
constexpr size_t OFF_YB = OFF_W2 + 1179648;         // y  bf16, NCHW (12.58M)
constexpr size_t OFF_YN = OFF_YB + 25165824;        // yn bf16, WMMA-A swizzled (12.58M)
constexpr size_t OFF_HB = OFF_YN + 25165824;        // h  bf16, WMMA-A swizzled (50.33M)

// ---------------------------------------------------------------------------
// K0: convert + swizzle fc weights ([N,K] f32 row-major) into the CDNA5 WMMA
// B-fragment layout: lane l holds N = ntile*16 + (l&15),
// K = ktile*32 + (l>>4)*8 + {0..7, 16..23}  (contiguous 32B per lane).
// ---------------------------------------------------------------------------
__global__ __launch_bounds__(256) void wswz_kernel(const float* __restrict__ w,
                                                   bf16* __restrict__ wsw,
                                                   int Kdim, int Ktiles) {
    const int idx = blockIdx.x * 256 + threadIdx.x;   // over N*K
    const int o = idx / Kdim;
    const int k = idx - o * Kdim;
    const int kr = k & 31;
    const int ln = (o & 15) + ((kr >> 3) & 1) * 16;
    const int e  = (kr & 7) + ((kr >> 4) & 1) * 8;
    wsw[(((size_t)(o >> 4) * Ktiles + (k >> 5)) * 32 + ln) * 16 + e] = (bf16)w[idx];
}

// ---------------------------------------------------------------------------
// K1: depthwise convs + identity; one block per (b,c) plane of 32x32.
// Writes y (bf16, NCHW) and deterministic per-block masked sums.
// ---------------------------------------------------------------------------
__global__ __launch_bounds__(256) void conv_stats_kernel(
    const float* __restrict__ x, const int* __restrict__ mask,
    const float* __restrict__ w_hw, const float* __restrict__ b_hw,
    const float* __restrict__ w_w,  const float* __restrict__ b_w,
    const float* __restrict__ w_h,  const float* __restrict__ b_h,
    bf16* __restrict__ yb, float* __restrict__ p1, float* __restrict__ p2) {
    const int bc = blockIdx.x;
    const int b = bc / kDim, c = bc - b * kDim;
    const float* xp = x + (size_t)bc * kHW;
    bf16* yp = yb + (size_t)bc * kHW;
    const int* mp = mask + b * 64;
    const int t = threadIdx.x;
    float ls1 = 0.f, ls2 = 0.f;
    for (int it = 0; it < 4; ++it) {
        const int hw = t + it * 256;
        const int h = hw >> 5, w = hw & 31;
        float y;
        if (c < kDim - 3 * kGC) {                       // identity channels
            y = xp[hw];
        } else if (c < kDim - 2 * kGC) {                // 3x3 depthwise, pad 1
            const int g = c - (kDim - 3 * kGC);
            float a = b_hw[g];
            #pragma unroll
            for (int di = -1; di <= 1; ++di)
                #pragma unroll
                for (int dj = -1; dj <= 1; ++dj) {
                    const int hh = h + di, ww = w + dj;
                    if (hh >= 0 && hh < 32 && ww >= 0 && ww < 32)
                        a += xp[hh * 32 + ww] * w_hw[g * 9 + (di + 1) * 3 + (dj + 1)];
                }
            y = a;
        } else if (c < kDim - kGC) {                    // 1x11 along W, pad 5
            const int g = c - (kDim - 2 * kGC);
            float a = b_w[g];
            #pragma unroll
            for (int dj = -5; dj <= 5; ++dj) {
                const int ww = w + dj;
                if (ww >= 0 && ww < 32) a += xp[h * 32 + ww] * w_w[g * 11 + dj + 5];
            }
            y = a;
        } else {                                        // 11x1 along H, pad 5
            const int g = c - (kDim - kGC);
            float a = b_h[g];
            #pragma unroll
            for (int di = -5; di <= 5; ++di) {
                const int hh = h + di;
                if (hh >= 0 && hh < 32) a += xp[hh * 32 + w] * w_h[g * 11 + di + 5];
            }
            y = a;
        }
        yp[hw] = (bf16)y;
        const float fm = (float)mp[(h >> 2) * 8 + (w >> 2)];
        ls1 += y * fm;
        ls2 += y * y * fm;
    }
    __shared__ float r1[256], r2[256];
    r1[t] = ls1; r2[t] = ls2;
    __syncthreads();
    for (int s = 128; s > 0; s >>= 1) {
        if (t < s) { r1[t] += r1[t + s]; r2[t] += r2[t + s]; }
        __syncthreads();
    }
    if (t == 0) { p1[bc] = r1[0]; p2[bc] = r2[0]; }
}

// ---------------------------------------------------------------------------
// K2: finalize BN stats (single block, 384 threads = one per channel).
// ---------------------------------------------------------------------------
__global__ void stats_finalize_kernel(const float* __restrict__ p1, const float* __restrict__ p2,
                                      const int* __restrict__ mask,
                                      const float* __restrict__ bn_w, const float* __restrict__ bn_b,
                                      float* __restrict__ sA, float* __restrict__ sB) {
    const int t = threadIdx.x;                  // 0..383
    __shared__ float red[384];
    __shared__ float cntS;
    float lc = 0.f;
    for (int i = t; i < 2048; i += 384) lc += (float)mask[i];
    red[t] = lc;
    __syncthreads();
    if (t == 0) {
        float s = 0.f;
        for (int i = 0; i < 384; ++i) s += red[i];
        cntS = s * 16.0f;                       // each 8x8 mask cell covers 4x4 pixels
    }
    __syncthreads();
    float s1 = 0.f, s2 = 0.f;
    for (int b = 0; b < 32; ++b) { s1 += p1[b * kDim + t]; s2 += p2[b * kDim + t]; }
    const float inv  = 1.0f / cntS;
    const float mean = s1 * inv;
    const float var  = s2 * inv - mean * mean;
    const float a    = bn_w[t] * rsqrtf(var + 1e-5f);
    sA[t] = a;
    sB[t] = bn_b[t] - mean * a;
}

// ---------------------------------------------------------------------------
// K3: yn = (a*y + b) * mask, scattered into the CDNA5 WMMA A-fragment layout
// (16-bit 16x32 A): lane l holds M = mtile*16 + (l&15),
// K = ktile*32 + (l>>4)*8 + {0..7, 16..23}  -> per-lane contiguous 32B chunk.
// ---------------------------------------------------------------------------
__global__ __launch_bounds__(256) void yn_swizzle_kernel(
    const bf16* __restrict__ yb, const int* __restrict__ mask,
    const float* __restrict__ sA, const float* __restrict__ sB,
    bf16* __restrict__ ynb) {
    const int idx = blockIdx.x * 256 + threadIdx.x;     // < 12582912, NCHW order
    const int b   = idx / 393216;                       // 384*1024
    const int rem = idx - b * 393216;
    const int c   = rem >> 10;
    const int hw  = rem & 1023;
    const int h = hw >> 5, w = hw & 31;
    const float fm = (float)mask[b * 64 + (h >> 2) * 8 + (w >> 2)];
    const float v  = (sA[c] * (float)yb[idx] + sB[c]) * fm;
    const int p  = b * 1024 + hw;                       // pixel row (M)
    const int kr = c & 31;
    const int ln = (p & 15) + ((kr >> 3) & 1) * 16;
    const int e  = (kr & 7) + ((kr >> 4) & 1) * 8;
    ynb[(((size_t)(p >> 4) * kKT1 + (c >> 5)) * 32 + ln) * 16 + e] = (bf16)v;
}

// ---------------------------------------------------------------------------
// K4: GEMM1 [32768x384] x [384x1536] via v_wmma_f32_16x16x32_bf16.
// 256 threads = 8 waves (4M x 2N); wave tile 32x64 (2x4 frags); block 128x128.
// Epilogue: +bias, exact GELU, write h in GEMM2's A-fragment layout.
// ---------------------------------------------------------------------------
__global__ __launch_bounds__(256) void gemm1_kernel(
    const bf16* __restrict__ ynb, const bf16* __restrict__ w1,
    const float* __restrict__ fc1_b, bf16* __restrict__ hb) {
    const int lane = threadIdx.x & 31;
    const int wv   = threadIdx.x >> 5;
    const int wm   = wv & 3, wn = wv >> 2;
    const int mt0  = blockIdx.x * 8 + wm * 2;           // m-tiles (16 rows each)
    const int nt0  = blockIdx.y * 8 + wn * 4;           // n-tiles
    const v8f vz = {0.f, 0.f, 0.f, 0.f, 0.f, 0.f, 0.f, 0.f};
    v8f acc[2][4];
    #pragma unroll
    for (int i = 0; i < 2; ++i)
        #pragma unroll
        for (int j = 0; j < 4; ++j) acc[i][j] = vz;

    for (int kt = 0; kt < kKT1; ++kt) {
        v16bf a[2], bm[4];
        #pragma unroll
        for (int i = 0; i < 2; ++i)
            a[i] = *(const v16bf*)(ynb + (((size_t)(mt0 + i) * kKT1 + kt) * 32 + lane) * 16);
        #pragma unroll
        for (int j = 0; j < 4; ++j)
            bm[j] = *(const v16bf*)(w1 + (((size_t)(nt0 + j) * kKT1 + kt) * 32 + lane) * 16);
        #pragma unroll
        for (int i = 0; i < 2; ++i)
            #pragma unroll
            for (int j = 0; j < 4; ++j)
                acc[i][j] = __builtin_amdgcn_wmma_f32_16x16x32_bf16(
                    false, a[i], false, bm[j], (short)0, acc[i][j], false, false);
    }

    const int half = lane >> 4;
    #pragma unroll
    for (int j = 0; j < 4; ++j) {
        const int o = (nt0 + j) * 16 + (lane & 15);
        const float bias = fc1_b[o];
        const int ktile = o >> 5, kr = o & 31;
        const int h2 = (kr >> 3) & 1;
        const int e  = (kr & 7) + ((kr >> 4) & 1) * 8;
        #pragma unroll
        for (int i = 0; i < 2; ++i) {
            #pragma unroll
            for (int r = 0; r < 8; ++r) {
                const int p = (mt0 + i) * 16 + half * 8 + r;    // C frag: M = r + half*8
                float v = acc[i][j][r] + bias;
                v = 0.5f * v * (1.0f + erff(v * 0.70710678118654752f));  // exact GELU
                const int ln = (p & 15) + h2 * 16;
                hb[(((size_t)(p >> 4) * kKT2 + ktile) * 32 + ln) * 16 + e] = (bf16)v;
            }
        }
    }
}

// ---------------------------------------------------------------------------
// K5: GEMM2 [32768x1536] x [1536x384] + bias, gamma, mask, residual -> NCHW out.
// ---------------------------------------------------------------------------
__global__ __launch_bounds__(256) void gemm2_kernel(
    const bf16* __restrict__ hb, const bf16* __restrict__ w2,
    const float* __restrict__ fc2_b, const float* __restrict__ gamma,
    const int* __restrict__ mask, const float* __restrict__ x,
    float* __restrict__ out) {
    const int lane = threadIdx.x & 31;
    const int wv   = threadIdx.x >> 5;
    const int wm   = wv & 3, wn = wv >> 2;
    const int mt0  = blockIdx.x * 8 + wm * 2;
    const int nt0  = blockIdx.y * 8 + wn * 4;
    const v8f vz = {0.f, 0.f, 0.f, 0.f, 0.f, 0.f, 0.f, 0.f};
    v8f acc[2][4];
    #pragma unroll
    for (int i = 0; i < 2; ++i)
        #pragma unroll
        for (int j = 0; j < 4; ++j) acc[i][j] = vz;

    for (int kt = 0; kt < kKT2; ++kt) {
        v16bf a[2], bm[4];
        #pragma unroll
        for (int i = 0; i < 2; ++i)
            a[i] = *(const v16bf*)(hb + (((size_t)(mt0 + i) * kKT2 + kt) * 32 + lane) * 16);
        #pragma unroll
        for (int j = 0; j < 4; ++j)
            bm[j] = *(const v16bf*)(w2 + (((size_t)(nt0 + j) * kKT2 + kt) * 32 + lane) * 16);
        #pragma unroll
        for (int i = 0; i < 2; ++i)
            #pragma unroll
            for (int j = 0; j < 4; ++j)
                acc[i][j] = __builtin_amdgcn_wmma_f32_16x16x32_bf16(
                    false, a[i], false, bm[j], (short)0, acc[i][j], false, false);
    }

    const int half = lane >> 4;
    #pragma unroll
    for (int j = 0; j < 4; ++j) {
        const int o = (nt0 + j) * 16 + (lane & 15);       // output channel
        const float bias = fc2_b[o];
        const float gam  = gamma[o];
        #pragma unroll
        for (int i = 0; i < 2; ++i) {
            #pragma unroll
            for (int r = 0; r < 8; ++r) {
                const int p  = (mt0 + i) * 16 + half * 8 + r;
                const int bb = p >> 10, hw = p & 1023;
                const int h = hw >> 5, w = hw & 31;
                const float fm = (float)mask[bb * 64 + (h >> 2) * 8 + (w >> 2)];
                const size_t xi = ((size_t)bb * kDim + o) * kHW + hw;
                out[xi] = x[xi] + gam * (acc[i][j][r] + bias) * fm;
            }
        }
    }
}

// ---------------------------------------------------------------------------
extern "C" void kernel_launch(void* const* d_in, const int* in_sizes, int n_in,
                              void* d_out, int out_size, void* d_ws, size_t ws_size,
                              hipStream_t stream) {
    (void)in_sizes; (void)n_in; (void)out_size; (void)ws_size;
    const float* x     = (const float*)d_in[0];
    const int*   mask  = (const int*)d_in[1];
    const float* w_hw  = (const float*)d_in[2];
    const float* b_hw  = (const float*)d_in[3];
    const float* w_w   = (const float*)d_in[4];
    const float* b_w   = (const float*)d_in[5];
    const float* w_h   = (const float*)d_in[6];
    const float* b_h   = (const float*)d_in[7];
    const float* bn_w  = (const float*)d_in[8];
    const float* bn_b  = (const float*)d_in[9];
    const float* fc1_w = (const float*)d_in[10];
    const float* fc1_b = (const float*)d_in[11];
    const float* fc2_w = (const float*)d_in[12];
    const float* fc2_b = (const float*)d_in[13];
    const float* gamma = (const float*)d_in[14];
    float* out = (float*)d_out;

    char* ws = (char*)d_ws;
    float* p1  = (float*)(ws + OFF_P1);
    float* p2  = (float*)(ws + OFF_P2);
    float* sA  = (float*)(ws + OFF_SA);
    float* sB  = (float*)(ws + OFF_SB);
    bf16* w1sw = (bf16*)(ws + OFF_W1);
    bf16* w2sw = (bf16*)(ws + OFF_W2);
    bf16* yb   = (bf16*)(ws + OFF_YB);
    bf16* ynb  = (bf16*)(ws + OFF_YN);
    bf16* hb   = (bf16*)(ws + OFF_HB);

    // 589824 weights each -> 2304 blocks of 256
    wswz_kernel<<<2304, 256, 0, stream>>>(fc1_w, w1sw, kDim, kKT1);
    wswz_kernel<<<2304, 256, 0, stream>>>(fc2_w, w2sw, kHid, kKT2);
    conv_stats_kernel<<<32 * kDim, 256, 0, stream>>>(x, mask, w_hw, b_hw, w_w, b_w,
                                                     w_h, b_h, yb, p1, p2);
    stats_finalize_kernel<<<1, 384, 0, stream>>>(p1, p2, mask, bn_w, bn_b, sA, sB);
    yn_swizzle_kernel<<<49152, 256, 0, stream>>>(yb, mask, sA, sB, ynb);
    gemm1_kernel<<<dim3(kNPix / 128, kHid / 128), 256, 0, stream>>>(ynb, w1sw, fc1_b, hb);
    gemm2_kernel<<<dim3(kNPix / 128, kDim / 128), 256, 0, stream>>>(hb, w2sw, fc2_b, gamma,
                                                                    mask, x, out);
}
